// MultiHeadAttention3d_11476152615350
// MI455X (gfx1250) — matrix operand order repaired and checked by reading, not compiled, not run on hardware
//
#include <hip/hip_runtime.h>
#include <hip/hip_bf16.h>

// ---------------------------------------------------------------------------
// MultiHeadAttention3d for MI455X (gfx1250, wave32, WMMA)
//   B=1, Cin=64, D=H=W=16 (S=4096), CK=CV=32, HEADS=2 -> N=8192, dk=dv=16
//   Split-K(x4) flash attention with v_wmma_f32_16x16x32_f16:
//     - 64-key blocks: 4 score WMMAs + 2 AV WMMAs, branch-free
//     - software-pipelined K/V prefetch hidden under base-2 softmax
//     - wave-private LDS, no hardware barriers in the hot loop
// ---------------------------------------------------------------------------

typedef __attribute__((ext_vector_type(16))) _Float16 v16h;
typedef __attribute__((ext_vector_type(8)))  _Float16 v8h;
typedef __attribute__((ext_vector_type(8)))  float    v8f;

union V16 { v16h v; v8h h[2]; _Float16 e[16]; };

#define S_SPATIAL 4096
#define NSEQ      8192     // S_SPATIAL * HEADS
#define DK        16
#define CIN       64
#define CKV       32
#define COUT      64
#define KPARTS    4
#define KEYS_PER_PART (NSEQ / KPARTS)      // 2048
#define LOG2E     1.4426950408889634f

// Wave-local LDS ordering: DS ops complete in order per wave (ISA 7.3);
// only compiler code motion must be fenced — zero hardware cost.
__device__ __forceinline__ void wave_lds_fence() {
    __builtin_amdgcn_sched_barrier(0);
    __builtin_amdgcn_wave_barrier();
    __builtin_amdgcn_sched_barrier(0);
}

// ---------------------------------------------------------------------------
// Kernel 1: QKV 1x1x1-conv projections.
//   x: [Cin, 4096] channel-major.  Writes:
//   qf  [8192][16] f16  (pre-scaled by dk^-0.5 * log2(e))    A-fragment rows
//   kfp [8256][32] f16  (cols 16..31 zero; rows 8192..8255 zero prefetch pad)
//   vt  [16][8192] f16  (transposed)                         V B-fragment rows
// ---------------------------------------------------------------------------
__global__ void __launch_bounds__(256) qkv_proj_kernel(
    const float* __restrict__ x,
    const float* __restrict__ qw, const float* __restrict__ qb,
    const float* __restrict__ kw, const float* __restrict__ kb,
    const float* __restrict__ vw, const float* __restrict__ vb,
    _Float16* __restrict__ qf, _Float16* __restrict__ kfp,
    _Float16* __restrict__ vt)
{
    int t = blockIdx.x * blockDim.x + threadIdx.x;   // 4096 * 32 threads
    int s = t >> 5;
    int o = t & 31;
    float aq = qb[o], ak = kb[o], av = vb[o];
    const float* wq = qw + o * CIN;
    const float* wk = kw + o * CIN;
    const float* wv = vw + o * CIN;
#pragma unroll 8
    for (int c = 0; c < CIN; ++c) {
        float xv = x[c * S_SPATIAL + s];
        aq += xv * wq[c];
        ak += xv * wk[c];
        av += xv * wv[c];
    }
    // N-flattening: n = s*HEADS + (o/16), head-dim index d = o%16
    int n = (s << 1) + (o >> 4);
    int d = o & 15;
    qf[n * DK + d]       = (_Float16)(aq * (0.25f * LOG2E));  // base-2 softmax
    kfp[n * 32 + d]      = (_Float16)ak;
    kfp[n * 32 + 16 + d] = (_Float16)0.0f;    // zero contraction pad
    vt[d * NSEQ + n]     = (_Float16)av;
    if (t < 64 * 32)                          // zero 64-row prefetch pad
        kfp[NSEQ * 32 + t] = (_Float16)0.0f;
}

// ---------------------------------------------------------------------------
// Kernel 2: split-K flash attention.  Wave w: qtile = w>>2, part = w&3.
// 32 iterations of 64 keys; emits unnormalized partials (O, m, l).
// ---------------------------------------------------------------------------
__global__ void __launch_bounds__(256) flash_attn_kernel(
    const _Float16* __restrict__ qf,
    const _Float16* __restrict__ kfp,
    const _Float16* __restrict__ vt,
    float* __restrict__ opart,     // [KPARTS][8192][16]
    float* __restrict__ mpart,     // [KPARTS][8192]
    float* __restrict__ lpart)     // [KPARTS][8192]
{
    __shared__ float    sS[8][16 * 64];     // per-wave score tile (C dump)
    __shared__ _Float16 sP[8][16 * 64];     // per-wave probs, row-major 16x64
    __shared__ float    sScale[8][16];      // per-row rescale broadcast

    const int lane = threadIdx.x & 31;
    const int wave = threadIdx.x >> 5;
    const int half = lane >> 4;
    const int lm   = lane & 15;
    const int w     = blockIdx.x * 8 + wave;   // 0..2047
    const int qtile = w >> 2;                  // 0..511
    const int part  = w & 3;                   // 0..3
    const int keybase = part * KEYS_PER_PART;

    v8f oacc; v8f zf;
#pragma unroll
    for (int i = 0; i < 8; ++i) { oacc[i] = 0.0f; zf[i] = 0.0f; }
    v8h zh;
#pragma unroll
    for (int i = 0; i < 8; ++i) zh[i] = (_Float16)0.0f;

    // Q A-fragment (16x32 f16, K=16..31 zero-padded).
    V16 qa;
    qa.h[0] = half ? zh : *(const v8h*)(qf + (qtile * 16 + lm) * DK);
    qa.h[1] = half ? zh : *(const v8h*)(qf + (qtile * 16 + lm) * DK + 8);

    // Running softmax state (base-2), identical in lanes lm and lm+16.
    float m_i = -1e30f;
    float l_i = 0.0f;

    float*    Swave = &sS[wave][0];
    _Float16* Pwave = &sP[wave][0];

    // K^T B-fragment base: lane col N = lm, VGPR v holds K = 2v + 16*half;
    // padded rows make all 4 loads unconditional contiguous 32B.
    const _Float16* kbase = kfp + (size_t)lm * 32 + half * 16;

    // ---- prologue: preload first K block ----------------------------------
    V16 kb0, kb1, kb2, kb3;
    kb0.v = *(const v16h*)(kbase + (size_t)(keybase +  0) * 32);
    kb1.v = *(const v16h*)(kbase + (size_t)(keybase + 16) * 32);
    kb2.v = *(const v16h*)(kbase + (size_t)(keybase + 32) * 32);
    kb3.v = *(const v16h*)(kbase + (size_t)(keybase + 48) * 32);

#pragma unroll 1
    for (int blk = 0; blk < KEYS_PER_PART / 64; ++blk) {
        const int key0 = keybase + blk * 64;

        // ---- scores: four 16x16 tiles covering 64 keys --------------------
        v8f s0 = __builtin_amdgcn_wmma_f32_16x16x32_f16(
            false, qa.v, false, kb0.v, (short)0, zf, false, false);
        v8f s1 = __builtin_amdgcn_wmma_f32_16x16x32_f16(
            false, qa.v, false, kb1.v, (short)0, zf, false, false);
        v8f s2 = __builtin_amdgcn_wmma_f32_16x16x32_f16(
            false, qa.v, false, kb2.v, (short)0, zf, false, false);
        v8f s3 = __builtin_amdgcn_wmma_f32_16x16x32_f16(
            false, qa.v, false, kb3.v, (short)0, zf, false, false);

        // ---- dump C-layout scores: row m = j + 8*half, col = g*16 + lm ----
        {
            float* Sd = Swave + (8 * half) * 64 + lm;
#pragma unroll
            for (int j = 0; j < 8; ++j) {
                Sd[j * 64 +  0] = s0[j];
                Sd[j * 64 + 16] = s1[j];
                Sd[j * 64 + 32] = s2[j];
                Sd[j * 64 + 48] = s3[j];
            }
        }
        wave_lds_fence();

        // ---- prefetch next K block + this block's V: latency hidden -------
        // under the softmax VALU segment (consumed only after next fence).
        kb0.v = *(const v16h*)(kbase + (size_t)(key0 + 64 +  0) * 32);
        kb1.v = *(const v16h*)(kbase + (size_t)(key0 + 64 + 16) * 32);
        kb2.v = *(const v16h*)(kbase + (size_t)(key0 + 64 + 32) * 32);
        kb3.v = *(const v16h*)(kbase + (size_t)(key0 + 64 + 48) * 32);
        V16 vb0, vb1;
        vb0.v = *(const v16h*)(vt + lm * NSEQ + key0 +      16 * half);
        vb1.v = *(const v16h*)(vt + lm * NSEQ + key0 + 32 + 16 * half);

        // ---- online softmax (base-2): lane (half,lm) owns 32 of 64 scores -
        const float* r = Swave + lm * 64 + half * 32;
        float pmax = -1e30f;
#pragma unroll
        for (int j = 0; j < 32; ++j) pmax = fmaxf(pmax, r[j]);
        pmax = fmaxf(pmax, __shfl_xor(pmax, 16, 32));   // combine lane pair
        float mx   = fmaxf(m_i, pmax);
        float corr = exp2f(m_i - mx);

        float p[32];
        float psum = 0.0f;
#pragma unroll
        for (int j = 0; j < 32; ++j) {
            p[j] = exp2f(r[j] - mx);
            psum += p[j];
        }
        psum += __shfl_xor(psum, 16, 32);               // combine lane pair
        l_i = l_i * corr + psum;
        m_i = mx;
        sScale[wave][lm] = corr;   // lanes lm & lm+16 store identical value

        // ---- store probs as f16, row-major 16x64 --------------------------
        {
            _Float16* Pd = Pwave + lm * 64 + half * 32;
#pragma unroll
            for (int c = 0; c < 4; ++c) {
                v8h ph;
#pragma unroll
                for (int i = 0; i < 8; ++i) ph[i] = (_Float16)p[c * 8 + i];
                *(v8h*)(Pd + c * 8) = ph;
            }
        }
        wave_lds_fence();

        // ---- rescale O accumulator (C layout: row = j + 8*half) -----------
#pragma unroll
        for (int j = 0; j < 8; ++j) oacc[j] *= sScale[wave][8 * half + j];

        // ---- P A-fragments (two 16x32 tiles, full contraction) ------------
        const _Float16* Pr = Pwave + lm * 64;
        V16 pa0, pa1;
        pa0.h[0] = *(const v8h*)(Pr +      half * 16);
        pa0.h[1] = *(const v8h*)(Pr +  8 + half * 16);
        pa1.h[0] = *(const v8h*)(Pr + 32 + half * 16);
        pa1.h[1] = *(const v8h*)(Pr + 40 + half * 16);

        // ---- O += P @ V ---------------------------------------------------
        oacc = __builtin_amdgcn_wmma_f32_16x16x32_f16(
            false, pa0.v, false, vb0.v, (short)0, oacc, false, false);
        oacc = __builtin_amdgcn_wmma_f32_16x16x32_f16(
            false, pa1.v, false, vb1.v, (short)0, oacc, false, false);

        wave_lds_fence();   // protect sS/sP against next iteration
    }

    // ---- emit unnormalized partials ---------------------------------------
    const int nrow = qtile * 16;
    mpart[part * NSEQ + nrow + lm] = m_i;   // both half-lanes: same value
    lpart[part * NSEQ + nrow + lm] = l_i;
#pragma unroll
    for (int j = 0; j < 8; ++j) {
        int row = j + 8 * half;
        opart[((size_t)part * NSEQ + nrow + row) * DK + lm] = oacc[j];
    }
}

// ---------------------------------------------------------------------------
// Kernel 3: split-K merge.  O = sum_p O_p*2^(m_p-m*) / sum_p l_p*2^(m_p-m*)
// ---------------------------------------------------------------------------
__global__ void __launch_bounds__(256) merge_kernel(
    const float* __restrict__ opart,
    const float* __restrict__ mpart,
    const float* __restrict__ lpart,
    float* __restrict__ attnout)            // [8192][16]
{
    int t = blockIdx.x * blockDim.x + threadIdx.x;   // 8192 * 16 threads
    int n = t >> 4;
    int d = t & 15;
    float m0 = mpart[0 * NSEQ + n], m1 = mpart[1 * NSEQ + n];
    float m2 = mpart[2 * NSEQ + n], m3 = mpart[3 * NSEQ + n];
    float ms = fmaxf(fmaxf(m0, m1), fmaxf(m2, m3));
    float w0 = exp2f(m0 - ms), w1 = exp2f(m1 - ms);
    float w2 = exp2f(m2 - ms), w3 = exp2f(m3 - ms);
    float lsum = lpart[0 * NSEQ + n] * w0 + lpart[1 * NSEQ + n] * w1 +
                 lpart[2 * NSEQ + n] * w2 + lpart[3 * NSEQ + n] * w3;
    float acc = opart[((size_t)0 * NSEQ + n) * DK + d] * w0 +
                opart[((size_t)1 * NSEQ + n) * DK + d] * w1 +
                opart[((size_t)2 * NSEQ + n) * DK + d] * w2 +
                opart[((size_t)3 * NSEQ + n) * DK + d] * w3;
    attnout[n * DK + d] = acc / lsum;
}

// ---------------------------------------------------------------------------
// Kernel 4: output 1x1x1 conv.  attnout flattened as [4096][32], y: [64][4096]
// ---------------------------------------------------------------------------
__global__ void __launch_bounds__(256) out_proj_kernel(
    const float* __restrict__ attnout,
    const float* __restrict__ ow, const float* __restrict__ ob,
    float* __restrict__ y)
{
    int t = blockIdx.x * blockDim.x + threadIdx.x;   // 4096 * 64 threads
    int s = t & (S_SPATIAL - 1);
    int o = t >> 12;
    const float* r = attnout + s * CKV;
    const float* w = ow + o * CKV;
    float acc = ob[o];
#pragma unroll
    for (int cv = 0; cv < CKV; ++cv) acc += r[cv] * w[cv];
    y[o * S_SPATIAL + s] = acc;
}

// ---------------------------------------------------------------------------
extern "C" void kernel_launch(void* const* d_in, const int* in_sizes, int n_in,
                              void* d_out, int out_size, void* d_ws, size_t ws_size,
                              hipStream_t stream) {
    const float* x  = (const float*)d_in[0];
    const float* qw = (const float*)d_in[1];
    const float* qb = (const float*)d_in[2];
    const float* kw = (const float*)d_in[3];
    const float* kb = (const float*)d_in[4];
    const float* vw = (const float*)d_in[5];
    const float* vb = (const float*)d_in[6];
    const float* ow = (const float*)d_in[7];
    const float* ob = (const float*)d_in[8];
    float* y = (float*)d_out;

    char* ws = (char*)d_ws;
    const size_t OFF_QF  = 0;                         // 256 KB
    const size_t OFF_KFP = OFF_QF  + (256u  << 10);   // 528 KB (padded K)
    const size_t OFF_VT  = OFF_KFP + (544u  << 10);   // 256 KB
    const size_t OFF_OP  = OFF_VT  + (256u  << 10);   // 2 MB  (O partials)
    const size_t OFF_M   = OFF_OP  + (2048u << 10);   // 128 KB
    const size_t OFF_L   = OFF_M   + (128u  << 10);   // 128 KB
    const size_t OFF_AO  = OFF_L   + (128u  << 10);   // 512 KB

    _Float16* qf      = (_Float16*)(ws + OFF_QF);
    _Float16* kfp     = (_Float16*)(ws + OFF_KFP);
    _Float16* vt      = (_Float16*)(ws + OFF_VT);
    float*    opart   = (float*)   (ws + OFF_OP);
    float*    mpart   = (float*)   (ws + OFF_M);
    float*    lpart   = (float*)   (ws + OFF_L);
    float*    attnout = (float*)   (ws + OFF_AO);

    // 1) QKV projections: 4096*32 threads
    qkv_proj_kernel<<<512, 256, 0, stream>>>(x, qw, qb, kw, kb, vw, vb,
                                             qf, kfp, vt);
    // 2) split-K flash attention: 512 Q tiles x 4 key partitions = 2048 waves
    flash_attn_kernel<<<256, 256, 0, stream>>>(qf, kfp, vt,
                                               opart, mpart, lpart);
    // 3) split-K merge: 8192*16 threads
    merge_kernel<<<512, 256, 0, stream>>>(opart, mpart, lpart, attnout);
    // 4) output projection: 4096*64 threads
    out_proj_kernel<<<1024, 256, 0, stream>>>(attnout, ow, ob, y);
}
